// Attention_45122926412037
// MI455X (gfx1250) — compile-verified, matrix operand order checked by reading, not run on hardware
//
#include <hip/hip_runtime.h>

// ---------------------------------------------------------------------------
// Problem constants
// ---------------------------------------------------------------------------
#define Bn 128
#define Ln 196
#define Fd 2048
#define Ad 1024
#define Hd 1024

#define MROWS  32           // L rows per block
#define MTILES 7            // ceil(196/32)
#define KTILES (Fd / 32)    // 64 k-tiles of 32
#define NTILES (Ad / 16)    // 64 n-tiles of 16

// LDS row stride for staged A tile (bf16 elems). 2048 + 24 pad -> 4144 bytes
// -> bank stride 12 (mod 64): optimal 2-way-only conflicts for b128 loads.
#define LSTR 2072

typedef __attribute__((ext_vector_type(16))) __bf16          v16bf;
typedef __attribute__((ext_vector_type(8)))  float           v8f;
typedef __attribute__((ext_vector_type(8)))  unsigned short  v8u16;
typedef __attribute__((ext_vector_type(16))) unsigned short  v16u16;

union FragU {
    v16u16 v;
    v8u16  h[2];
};

__device__ __forceinline__ unsigned short f2bf(float f) {
    union { float f; unsigned int u; } x;
    x.f = f;
    unsigned int r = x.u + 0x7FFFu + ((x.u >> 16) & 1u);   // RNE
    return (unsigned short)(r >> 16);
}

// ---------------------------------------------------------------------------
// Kernel 1: repack W_enc (F x A, fp32 row-major) into bf16 WMMA B-fragments.
// Fragment (nt, kt) = 32x16 tile of W_enc rows [kt*32, +32), cols [nt*16,+16).
// Layout per ISA 7.12.2 (16-bit B 32x16): lane n (0-15) = col n holds K 0-7 /
// 16-23; lane n+16 = col n holds K 8-15 / 24-31.  Stored lane-major, 32B/lane.
// packW index (ushort): ((nt*KTILES + kt)*32 + lane)*16
// ---------------------------------------------------------------------------
__global__ __launch_bounds__(256) void pack_wenc_kernel(
    const float* __restrict__ W_enc, unsigned short* __restrict__ packW)
{
    int gid  = blockIdx.x * 256 + threadIdx.x;   // 4096 frags * 32 lanes
    int frag = gid >> 5;
    int lane = gid & 31;
    int nt   = frag >> 6;        // / KTILES
    int kt   = frag & 63;        // % KTILES
    int col  = nt * 16 + (lane & 15);
    int kb   = kt * 32 + ((lane & 16) ? 8 : 0);

    unsigned short* dst = packW + (size_t)frag * 512 + (size_t)lane * 16;
#pragma unroll
    for (int j = 0; j < 8; ++j)
        dst[j] = f2bf(W_enc[(size_t)(kb + j) * Ad + col]);
#pragma unroll
    for (int j = 0; j < 8; ++j)
        dst[8 + j] = f2bf(W_enc[(size_t)(kb + 16 + j) * Ad + col]);
}

// ---------------------------------------------------------------------------
// Kernel 2: attn2b[b][a] = hidden[b] . W_hid[:,a] + b_hid[a] + b_enc[a]
// ---------------------------------------------------------------------------
__global__ __launch_bounds__(256) void attn2_kernel(
    const float* __restrict__ hidden, const float* __restrict__ W_hid,
    const float* __restrict__ b_hid,  const float* __restrict__ b_enc,
    float* __restrict__ attn2b)
{
    int b = blockIdx.x, t = threadIdx.x;
    __shared__ float sh[Hd];
    for (int i = t; i < Hd; i += 256) sh[i] = hidden[(size_t)b * Hd + i];
    __syncthreads();

    float a0 = 0.f, a1 = 0.f, a2 = 0.f, a3 = 0.f;
    for (int h = 0; h < Hd; ++h) {
        float hv = sh[h];
        const float* wr = W_hid + (size_t)h * Ad;
        a0 += hv * wr[t];
        a1 += hv * wr[t + 256];
        a2 += hv * wr[t + 512];
        a3 += hv * wr[t + 768];
    }
    float* o = attn2b + (size_t)b * Ad;
    o[t]       = a0 + b_hid[t]       + b_enc[t];
    o[t + 256] = a1 + b_hid[t + 256] + b_enc[t + 256];
    o[t + 512] = a2 + b_hid[t + 512] + b_enc[t + 512];
    o[t + 768] = a3 + b_hid[t + 768] + b_enc[t + 768];
}

// ---------------------------------------------------------------------------
// Kernel 3 (main): per (b, 32-row L tile) block computes
//   scores[b][l] = W_full . relu(enc[b,l,:] @ W_enc + attn2b[b,:]) + b_full
// bf16 WMMA, fp32 accumulation.  8 waves; each wave owns 8 n-tiles and BOTH
// 16-row M-tiles, so every B fragment load feeds 2 WMMAs (halved L2 traffic).
// ---------------------------------------------------------------------------
__global__ __launch_bounds__(256) void attn_scores_kernel(
    const float* __restrict__ enc, const unsigned short* __restrict__ packW,
    const float* __restrict__ attn2b, const float* __restrict__ W_full,
    const float* __restrict__ b_full, float* __restrict__ scores)
{
    __shared__ __align__(16) unsigned short ldsA[MROWS * LSTR];   // ~132 KB
    __shared__ float sScore[MROWS];

    const int tid = threadIdx.x;
    const int b   = blockIdx.y;
    const int l0  = blockIdx.x * MROWS;

    if (tid < MROWS) sScore[tid] = 0.f;

    // ---- stage A tile (32 x 2048 fp32 -> bf16 in LDS), zero-fill tail rows
    const float* encB = enc + ((size_t)b * Ln + l0) * Fd;
    for (int idx = tid; idx < MROWS * (Fd / 4); idx += 256) {
        int row = idx / (Fd / 4);
        int c4  = (idx % (Fd / 4)) * 4;
        float4 v = make_float4(0.f, 0.f, 0.f, 0.f);
        if (l0 + row < Ln) v = *(const float4*)(encB + (size_t)row * Fd + c4);
        unsigned short* d = &ldsA[row * LSTR + c4];
        d[0] = f2bf(v.x); d[1] = f2bf(v.y); d[2] = f2bf(v.z); d[3] = f2bf(v.w);
    }
    __syncthreads();

    const int lane = tid & 31;
    const int wave = tid >> 5;
    const int row  = lane & 15;
    const int hi   = lane >> 4;                 // half-wave select
    // A-fragment base (ISA 7.12.2, 16-bit A 16x32): lanes 0-15 hold K 0-7 /
    // 16-23 of the k-tile; lanes 16-31 hold K 8-15 / 24-31.
    const unsigned short* aPtr = &ldsA[row * LSTR + hi * 8];

    const float* a2p = attn2b + (size_t)b * Ad;

    // persistent per-lane row accumulators (score partials for 2 M-tiles)
    float racc0[8], racc1[8];
#pragma unroll
    for (int j = 0; j < 8; ++j) { racc0[j] = 0.f; racc1[j] = 0.f; }

    for (int p = 0; p < 2; ++p) {
        const int ntBase = wave * 8 + p * 4;    // 4 n-tiles per pass
        v8f acc0[4], acc1[4];                   // [n-tile] for M-tile 0 / 1
#pragma unroll
        for (int q = 0; q < 4; ++q) {
            acc0[q] = (v8f){0.f, 0.f, 0.f, 0.f, 0.f, 0.f, 0.f, 0.f};
            acc1[q] = (v8f){0.f, 0.f, 0.f, 0.f, 0.f, 0.f, 0.f, 0.f};
        }

        const unsigned short* bp[4];
#pragma unroll
        for (int q = 0; q < 4; ++q)
            bp[q] = packW + ((size_t)(ntBase + q) * KTILES) * 512
                          + (size_t)lane * 16;

#pragma unroll 2
        for (int kt = 0; kt < KTILES; ++kt) {
            FragU a0u, a1u;
            a0u.h[0] = *(const v8u16*)(aPtr + kt * 32);
            a0u.h[1] = *(const v8u16*)(aPtr + kt * 32 + 16);
            a1u.h[0] = *(const v8u16*)(aPtr + 16 * LSTR + kt * 32);
            a1u.h[1] = *(const v8u16*)(aPtr + 16 * LSTR + kt * 32 + 16);
            v16bf af0 = __builtin_bit_cast(v16bf, a0u.v);
            v16bf af1 = __builtin_bit_cast(v16bf, a1u.v);
#pragma unroll
            for (int q = 0; q < 4; ++q) {
                v16u16 bv = *(const v16u16*)(bp[q] + (size_t)kt * 512);
                v16bf  bf = __builtin_bit_cast(v16bf, bv);
                acc0[q] = __builtin_amdgcn_wmma_f32_16x16x32_bf16(
                    false, af0, false, bf, (short)0, acc0[q], false, false);
                acc1[q] = __builtin_amdgcn_wmma_f32_16x16x32_bf16(
                    false, af1, false, bf, (short)0, acc1[q], false, false);
            }
        }

        // ---- fused epilogue: +attn2b, relu, *W_full into row accumulators
#pragma unroll
        for (int q = 0; q < 4; ++q) {
            int col   = (ntBase + q) * 16 + (lane & 15);
            float wf  = W_full[col];
            float add = a2p[col];
#pragma unroll
            for (int j = 0; j < 8; ++j) {
                float v0 = acc0[q][j] + add;
                racc0[j] += wf * (v0 > 0.f ? v0 : 0.f);
                float v1 = acc1[q][j] + add;
                racc1[j] += wf * (v1 > 0.f ? v1 : 0.f);
            }
        }
    }

    // ---- single cross-lane reduction at the end (16-lane groups)
#pragma unroll
    for (int j = 0; j < 8; ++j) {
        float v0 = racc0[j];
        float v1 = racc1[j];
        for (int off = 1; off < 16; off <<= 1) {
            v0 += __shfl_xor(v0, off, 32);
            v1 += __shfl_xor(v1, off, 32);
        }
        if ((lane & 15) == 0) {
            atomicAdd(&sScore[hi * 8 + j],      v0);   // ds_add_f32
            atomicAdd(&sScore[16 + hi * 8 + j], v1);
        }
    }

    __syncthreads();
    if (tid < MROWS && (l0 + tid) < Ln)
        scores[b * Ln + l0 + tid] = sScore[tid] + b_full[0];
}

// ---------------------------------------------------------------------------
// Kernel 4: softmax over L per batch
// ---------------------------------------------------------------------------
__global__ __launch_bounds__(256) void softmax_kernel(
    const float* __restrict__ scores, float* __restrict__ alpha)
{
    int b = blockIdx.x, t = threadIdx.x;
    __shared__ float buf[256];
    float val = (t < Ln) ? scores[b * Ln + t] : -3.4e38f;
    buf[t] = val;
    __syncthreads();
    for (int s = 128; s > 0; s >>= 1) {
        if (t < s) buf[t] = fmaxf(buf[t], buf[t + s]);
        __syncthreads();
    }
    float mx = buf[0];
    __syncthreads();
    float e = (t < Ln) ? expf(val - mx) : 0.f;
    buf[t] = e;
    __syncthreads();
    for (int s = 128; s > 0; s >>= 1) {
        if (t < s) buf[t] += buf[t + s];
        __syncthreads();
    }
    float inv = 1.0f / buf[0];
    if (t < Ln) alpha[b * Ln + t] = e * inv;
}

// ---------------------------------------------------------------------------
// Kernel 5: context[b][f] = sum_l alpha[b][l] * enc[b][l][f]   (BW-bound)
// ---------------------------------------------------------------------------
__global__ __launch_bounds__(256) void context_kernel(
    const float* __restrict__ enc, const float* __restrict__ alpha,
    float* __restrict__ ctx)
{
    int b = blockIdx.y;
    int f = blockIdx.x * 256 + threadIdx.x;
    __shared__ float sa[Ln];
    for (int i = threadIdx.x; i < Ln; i += 256) sa[i] = alpha[b * Ln + i];
    __syncthreads();

    const float* e = enc + (size_t)b * Ln * Fd + f;
    float acc = 0.f;
#pragma unroll 4
    for (int l = 0; l < Ln; ++l)
        acc += sa[l] * e[(size_t)l * Fd];
    ctx[(size_t)b * Fd + f] = acc;
}

// ---------------------------------------------------------------------------
// Host-side launch
// ---------------------------------------------------------------------------
extern "C" void kernel_launch(void* const* d_in, const int* in_sizes, int n_in,
                              void* d_out, int out_size, void* d_ws, size_t ws_size,
                              hipStream_t stream)
{
    const float* enc    = (const float*)d_in[0];   // (B, L, F)
    const float* hidden = (const float*)d_in[1];   // (B, H)
    const float* W_enc  = (const float*)d_in[2];   // (F, A)
    const float* b_enc  = (const float*)d_in[3];   // (A,)
    const float* W_hid  = (const float*)d_in[4];   // (H, A)
    const float* b_hid  = (const float*)d_in[5];   // (A,)
    const float* W_full = (const float*)d_in[6];   // (A,)
    const float* b_full = (const float*)d_in[7];   // (1,)

    float* ctx   = (float*)d_out;                       // B*F
    float* alpha = (float*)d_out + (size_t)Bn * Fd;     // B*L

    // workspace: packed W_enc (bf16) | attn2b fp32 | scores fp32  (~4.6 MB)
    unsigned short* packW = (unsigned short*)d_ws;
    float* attn2b = (float*)((char*)d_ws + (size_t)Fd * Ad * sizeof(unsigned short));
    float* scores = attn2b + (size_t)Bn * Ad;

    pack_wenc_kernel<<<(NTILES * KTILES * 32) / 256, 256, 0, stream>>>(W_enc, packW);
    attn2_kernel<<<Bn, 256, 0, stream>>>(hidden, W_hid, b_hid, b_enc, attn2b);

    dim3 gs(MTILES, Bn);
    attn_scores_kernel<<<gs, 256, 0, stream>>>(enc, packW, attn2b, W_full, b_full, scores);

    softmax_kernel<<<Bn, 256, 0, stream>>>(scores, alpha);

    dim3 gc(Fd / 256, Bn);
    context_kernel<<<gc, 256, 0, stream>>>(enc, alpha, ctx);
}
